// Attention_56487409877769
// MI455X (gfx1250) — matrix operand order repaired and checked by reading, not compiled, not run on hardware
//
#include <hip/hip_runtime.h>
#include <hip/hip_bf16.h>
#include <math.h>

// Problem constants (match reference)
#define BSZ  4
#define SEQ  1024
#define CH   1024
#define NH   16
#define HDIM 64
#define C3   3072
#define ATTN_SCALE 0.125f   // 64^-0.5

typedef __attribute__((ext_vector_type(16))) __bf16 v16bf;
typedef __attribute__((ext_vector_type(8)))  float  v8f;

union Frag  { v16bf v; uint4 q[2]; };
union Pack8 { unsigned short s[8]; uint4 q; };

__device__ __forceinline__ unsigned short f2bf(float f) {
  unsigned u = __builtin_bit_cast(unsigned, f);
  unsigned r = u + 0x7FFFu + ((u >> 16) & 1u);   // round-to-nearest-even
  return (unsigned short)(r >> 16);
}

__device__ __forceinline__ v8f wmma_bf16(const Frag& a, const Frag& b, v8f c) {
  return __builtin_amdgcn_wmma_f32_16x16x32_bf16(false, a.v, false, b.v,
                                                 (short)0, c, false, false);
}

// Async 16B global -> LDS copy (per lane), tracked with ASYNCcnt.
__device__ __forceinline__ void async_g2l_b128(unsigned lds_off, const void* gptr) {
  asm volatile("global_load_async_to_lds_b128 %0, %1, off"
               :: "v"(lds_off), "v"(gptr) : "memory");
}

// ---------------------------------------------------------------------------
// fp32 -> bf16 elementwise convert (weights)
// ---------------------------------------------------------------------------
__global__ void cvt_f32_bf16(const float* __restrict__ in,
                             unsigned short* __restrict__ out, int n) {
  int i = blockIdx.x * 256 + threadIdx.x;
  if (i < n) out[i] = f2bf(in[i]);
}

// ---------------------------------------------------------------------------
// Transpose-convert x (B,N,C) f32 -> xT (B,C,N) bf16
// ---------------------------------------------------------------------------
__global__ void tr_f32_bf16(const float* __restrict__ in,
                            unsigned short* __restrict__ out,
                            int rows, int cols) {
  __shared__ float tile[32][33];
  int b = blockIdx.z;
  const float* ip = in + (long long)b * rows * cols;
  unsigned short* op = out + (long long)b * rows * cols;
  int c0 = blockIdx.x * 32, r0 = blockIdx.y * 32;
  int tx = threadIdx.x, ty = threadIdx.y;
#pragma unroll
  for (int i = 0; i < 4; i++)
    tile[ty + i * 8][tx] = ip[(long long)(r0 + ty + i * 8) * cols + c0 + tx];
  __syncthreads();
#pragma unroll
  for (int i = 0; i < 4; i++)
    op[(long long)(c0 + ty + i * 8) * rows + r0 + tx] = f2bf(tile[tx][ty + i * 8]);
}

// ---------------------------------------------------------------------------
// Transpose V slice of QKV bf16: VT[b][c'][j] = QKV[b][j][2C + c']
// ---------------------------------------------------------------------------
__global__ void tr_v_bf16(const unsigned short* __restrict__ qkv,
                          unsigned short* __restrict__ vt) {
  __shared__ unsigned short tile[32][33];
  int b = blockIdx.z;
  const unsigned short* ip = qkv + (long long)b * SEQ * C3 + 2 * CH;
  unsigned short* op = vt + (long long)b * CH * SEQ;
  int c0 = blockIdx.x * 32, r0 = blockIdx.y * 32;
  int tx = threadIdx.x, ty = threadIdx.y;
#pragma unroll
  for (int i = 0; i < 4; i++)
    tile[ty + i * 8][tx] = ip[(long long)(r0 + ty + i * 8) * C3 + c0 + tx];
  __syncthreads();
#pragma unroll
  for (int i = 0; i < 4; i++)
    op[(long long)(c0 + ty + i * 8) * SEQ + r0 + tx] = tile[tx][ty + i * 8];
}

// ---------------------------------------------------------------------------
// WMMA GEMM:  C[m,n] = sum_k A[m,k] * B[n,k]  (+ bias_row[m] + bias_col[n])
// A: (M x K) bf16 row-major, B: (Nout x K) bf16 row-major (both K-contiguous)
// 8 waves per WG; each wave computes a 16(M) x 64(N) tile.
// B tile (64x32 bf16 = 4KB) double-buffered in LDS via async global->LDS DMA:
// tile i+1 streams in (ASYNCcnt) while tile i feeds the WMMAs.
// ---------------------------------------------------------------------------
__global__ __launch_bounds__(256) void gemm_bf16(
    const unsigned short* __restrict__ A, int lda, long long strideA,
    const unsigned short* __restrict__ Bm, int ldb, long long strideB,
    void* __restrict__ Cp, int ldc, long long strideC, int c_is_f32,
    const float* __restrict__ bias_row, const float* __restrict__ bias_col,
    int K) {
  __shared__ unsigned char btile[2 * 4096];   // 2 x (64 n-rows x 32 k bf16)
  int tid = threadIdx.x;
  int lane = tid & 31, wv = tid >> 5;
  int hl = lane >> 4, lr = lane & 15;
  int bidx = blockIdx.z;
  const unsigned short* Ab = A + bidx * strideA;
  const unsigned short* Bb = Bm + bidx * strideB;
  int m0 = blockIdx.x * 128 + wv * 16;
  int n0 = blockIdx.y * 64;

  // per-thread staging slot: 16B of the 4KB tile
  int srow = tid >> 2, schunk = tid & 3;
  const unsigned short* gstage = Bb + (long long)(n0 + srow) * ldb + schunk * 8;
  // low 32 bits of a generic __shared__ pointer == LDS byte offset (aperture in hi bits)
  unsigned lds_stage = (unsigned)(size_t)(btile + srow * 64 + schunk * 16);

  v8f acc[4];
#pragma unroll
  for (int t = 0; t < 4; t++)
#pragma unroll
    for (int v = 0; v < 8; v++) acc[t][v] = 0.0f;

  // prologue: stage tile 0 -> buffer 0
  async_g2l_b128(lds_stage, gstage);

  int nk = K >> 5;
  for (int i = 0; i < nk; i++) {
    __syncthreads();   // all waves done reading buffer (i+1)&1 (from iter i-1)
    if (i + 1 < nk) {
      async_g2l_b128(lds_stage + ((unsigned)((i + 1) & 1) << 12),
                     gstage + (i + 1) * 32);
      // async loads complete in order: <=1 outstanding => tile i is in LDS
      asm volatile("s_wait_asynccnt 1" ::: "memory");
    } else {
      asm volatile("s_wait_asynccnt 0" ::: "memory");
    }
    __syncthreads();   // tile i visible to all waves

    const unsigned char* bufp = btile + (i & 1) * 4096;
    Frag a;
    const unsigned short* ap = Ab + (long long)(m0 + lr) * lda + i * 32 + hl * 8;
    a.q[0] = *(const uint4*)(ap);
    a.q[1] = *(const uint4*)(ap + 16);

#pragma unroll
    for (int t = 0; t < 4; t++) {
      Frag bf_;
      const unsigned char* bp = bufp + (t * 16 + lr) * 64 + hl * 16;
      bf_.q[0] = *(const uint4*)(bp);
      bf_.q[1] = *(const uint4*)(bp + 32);
      acc[t] = wmma_bf16(a, bf_, acc[t]);
    }
  }

  float brow[8];
#pragma unroll
  for (int v = 0; v < 8; v++)
    brow[v] = bias_row ? bias_row[m0 + hl * 8 + v] : 0.0f;

#pragma unroll
  for (int t = 0; t < 4; t++) {
    int col = n0 + t * 16 + lr;
    float bc = bias_col ? bias_col[col] : 0.0f;
#pragma unroll
    for (int v = 0; v < 8; v++) {
      int row = m0 + hl * 8 + v;
      float val = acc[t][v] + brow[v] + bc;
      if (c_is_f32)
        ((float*)Cp)[bidx * strideC + (long long)row * ldc + col] = val;
      else
        ((unsigned short*)Cp)[bidx * strideC + (long long)row * ldc + col] = f2bf(val);
    }
  }
}

// ---------------------------------------------------------------------------
// Fused flash-attention per (b, h, 64-row block); 4 waves x 16 q-rows each.
// Output written transposed: aot[b][h*64+e][i] (bf16) for the next seq-linear.
// ---------------------------------------------------------------------------
__global__ __launch_bounds__(128) void attn_fused(
    const unsigned short* __restrict__ qkv,
    const unsigned short* __restrict__ vt,
    unsigned short* __restrict__ aot) {
  __shared__ unsigned char ptile[4 * 1024];   // per-wave 16x32 bf16 P staging
  int tid = threadIdx.x;
  int lane = tid & 31, wv = tid >> 5;
  int hl = lane >> 4, lr = lane & 15;
  int b = blockIdx.z, h = blockIdx.y;
  int i0 = blockIdx.x * 64 + wv * 16;

  const unsigned short* Qb = qkv + (long long)b * SEQ * C3 + h * HDIM;
  const unsigned short* Kb = Qb + CH;
  const unsigned short* Vb = vt + ((long long)b * CH + h * HDIM) * SEQ;
  unsigned char* pw = ptile + wv * 1024;

  // Q A-fragments (16 rows x 64 e) : two k=32 fragments, loaded once
  Frag qa0, qa1;
  {
    const unsigned short* qp = Qb + (long long)(i0 + lr) * C3 + hl * 8;
    qa0.q[0] = *(const uint4*)(qp);
    qa0.q[1] = *(const uint4*)(qp + 16);
    qa1.q[0] = *(const uint4*)(qp + 32);
    qa1.q[1] = *(const uint4*)(qp + 48);
  }

  v8f o[4];
  float mi[8], li[8];
#pragma unroll
  for (int t = 0; t < 4; t++)
#pragma unroll
    for (int v = 0; v < 8; v++) o[t][v] = 0.0f;
#pragma unroll
  for (int v = 0; v < 8; v++) { mi[v] = -3.0e38f; li[v] = 0.0f; }

  for (int jc = 0; jc < SEQ; jc += 32) {
    v8f s0, s1;
#pragma unroll
    for (int v = 0; v < 8; v++) { s0[v] = 0.0f; s1[v] = 0.0f; }

    {   // S tile for keys [jc, jc+16)
      const unsigned short* kp = Kb + (long long)(jc + lr) * C3 + hl * 8;
      Frag kb0, kb1;
      kb0.q[0] = *(const uint4*)(kp);       kb0.q[1] = *(const uint4*)(kp + 16);
      kb1.q[0] = *(const uint4*)(kp + 32);  kb1.q[1] = *(const uint4*)(kp + 48);
      s0 = wmma_bf16(qa0, kb0, s0);
      s0 = wmma_bf16(qa1, kb1, s0);
    }
    {   // S tile for keys [jc+16, jc+32)
      const unsigned short* kp = Kb + (long long)(jc + 16 + lr) * C3 + hl * 8;
      Frag kb0, kb1;
      kb0.q[0] = *(const uint4*)(kp);       kb0.q[1] = *(const uint4*)(kp + 16);
      kb1.q[0] = *(const uint4*)(kp + 32);  kb1.q[1] = *(const uint4*)(kp + 48);
      s1 = wmma_bf16(qa0, kb0, s1);
      s1 = wmma_bf16(qa1, kb1, s1);
    }

    // Online softmax: rows live in (v, half-wave); reduce across 16 lanes.
    float scl[8];
#pragma unroll
    for (int v = 0; v < 8; v++) {
      s0[v] *= ATTN_SCALE; s1[v] *= ATTN_SCALE;
      float cm = fmaxf(s0[v], s1[v]);
#pragma unroll
      for (int off = 1; off < 16; off <<= 1) cm = fmaxf(cm, __shfl_xor(cm, off, 16));
      float mn = fmaxf(mi[v], cm);
      float sc = __expf(mi[v] - mn);
      float p0 = __expf(s0[v] - mn);
      float p1 = __expf(s1[v] - mn);
      float rs = p0 + p1;
#pragma unroll
      for (int off = 1; off < 16; off <<= 1) rs += __shfl_xor(rs, off, 16);
      li[v] = li[v] * sc + rs;
      mi[v] = mn;
      scl[v] = sc;
      s0[v] = p0; s1[v] = p1;
    }
#pragma unroll
    for (int t = 0; t < 4; t++)
#pragma unroll
      for (int v = 0; v < 8; v++) o[t][v] *= scl[v];

    // P (D-layout f32) -> wave-private LDS as 16x32 bf16 tile
#pragma unroll
    for (int v = 0; v < 8; v++) {
      int row = v + 8 * hl;
      *(unsigned short*)(pw + row * 64 + lr * 2)      = f2bf(s0[v]);
      *(unsigned short*)(pw + row * 64 + 32 + lr * 2) = f2bf(s1[v]);
    }
    asm volatile("s_wait_dscnt 0" ::: "memory");   // wave-private RAW fence

    Frag pa;   // P as A-fragment (16 x 32)
    pa.q[0] = *(const uint4*)(pw + lr * 64 + hl * 16);
    pa.q[1] = *(const uint4*)(pw + lr * 64 + 32 + hl * 16);

#pragma unroll
    for (int t = 0; t < 4; t++) {   // O += P * V  (VT is K-contiguous)
      const unsigned short* vp = Vb + (long long)(t * 16 + lr) * SEQ + jc + hl * 8;
      Frag vb_;
      vb_.q[0] = *(const uint4*)(vp);
      vb_.q[1] = *(const uint4*)(vp + 16);
      o[t] = wmma_bf16(pa, vb_, o[t]);
    }
  }

  float inv[8];
#pragma unroll
  for (int v = 0; v < 8; v++) inv[v] = 1.0f / li[v];

#pragma unroll
  for (int t = 0; t < 4; t++) {   // transposed store: 8 contiguous bf16 / lane
    int e = t * 16 + lr;
    Pack8 pk;
#pragma unroll
    for (int v = 0; v < 8; v++) pk.s[v] = f2bf(o[t][v] * inv[v]);
    unsigned short* op = aot + ((long long)b * CH + h * HDIM + e) * SEQ + i0 + hl * 8;
    *(uint4*)op = pk.q;
  }
}

// ---------------------------------------------------------------------------
extern "C" void kernel_launch(void* const* d_in, const int* in_sizes, int n_in,
                              void* d_out, int out_size, void* d_ws, size_t ws_size,
                              hipStream_t stream) {
  const float* x   = (const float*)d_in[0];
  const float* Wq0 = (const float*)d_in[1];
  const float* bq0 = (const float*)d_in[2];
  const float* Wq1 = (const float*)d_in[3];
  const float* bq1 = (const float*)d_in[4];
  const float* Wp0 = (const float*)d_in[5];
  const float* bp0 = (const float*)d_in[6];
  const float* Wp1 = (const float*)d_in[7];
  const float* bp1 = (const float*)d_in[8];
  float* out = (float*)d_out;

  char* ws = (char*)d_ws;
  size_t off = 0;
  auto alloc = [&](size_t bytes) -> void* {
    void* p = ws + off;
    off += (bytes + 255) & ~(size_t)255;
    return p;
  };
  unsigned short* xT   = (unsigned short*)alloc((size_t)BSZ * CH * SEQ * 2);
  unsigned short* Wq0b = (unsigned short*)alloc((size_t)SEQ * SEQ * 2);
  unsigned short* Wq1b = (unsigned short*)alloc((size_t)C3 * CH * 2);
  unsigned short* Wp0b = (unsigned short*)alloc((size_t)SEQ * SEQ * 2);
  unsigned short* Wp1b = (unsigned short*)alloc((size_t)CH * CH * 2);
  unsigned short* Y1   = (unsigned short*)alloc((size_t)BSZ * SEQ * CH * 2);
  unsigned short* QKV  = (unsigned short*)alloc((size_t)BSZ * SEQ * C3 * 2);
  unsigned short* VT   = (unsigned short*)alloc((size_t)BSZ * CH * SEQ * 2);
  unsigned short* AOT  = (unsigned short*)alloc((size_t)BSZ * CH * SEQ * 2);
  unsigned short* OUT2 = (unsigned short*)alloc((size_t)BSZ * SEQ * CH * 2);

  // 1) weights fp32 -> bf16
  cvt_f32_bf16<<<(SEQ * SEQ + 255) / 256, 256, 0, stream>>>(Wq0, Wq0b, SEQ * SEQ);
  cvt_f32_bf16<<<(C3 * CH + 255) / 256, 256, 0, stream>>>(Wq1, Wq1b, C3 * CH);
  cvt_f32_bf16<<<(SEQ * SEQ + 255) / 256, 256, 0, stream>>>(Wp0, Wp0b, SEQ * SEQ);
  cvt_f32_bf16<<<(CH * CH + 255) / 256, 256, 0, stream>>>(Wp1, Wp1b, CH * CH);

  // 2) xT[b][c][n] = bf16(x[b][n][c])
  tr_f32_bf16<<<dim3(CH / 32, SEQ / 32, BSZ), dim3(32, 8), 0, stream>>>(x, xT, SEQ, CH);

  // 3) seq linear: Y1[b][m][c] = sum_n Wq0[m,n]*x[b,n,c] + bq0[m]
  gemm_bf16<<<dim3(SEQ / 128, CH / 64, BSZ), 256, 0, stream>>>(
      Wq0b, SEQ, 0, xT, SEQ, (long long)CH * SEQ,
      Y1, CH, (long long)SEQ * CH, 0, bq0, nullptr, SEQ);

  // 4) channel linear: QKV[b][n][d] = sum_c Y1[b,n,c]*Wq1[d,c] + bq1[d]
  gemm_bf16<<<dim3(SEQ / 128, C3 / 64, BSZ), 256, 0, stream>>>(
      Y1, CH, (long long)SEQ * CH, Wq1b, CH, 0,
      QKV, C3, (long long)SEQ * C3, 0, nullptr, bq1, CH);

  // 5) VT[b][h*64+e][j] = V[b][j][h*64+e]
  tr_v_bf16<<<dim3(CH / 32, SEQ / 32, BSZ), dim3(32, 8), 0, stream>>>(QKV, VT);

  // 6) fused flash attention -> AOT (transposed layout)
  attn_fused<<<dim3(SEQ / 64, NH, BSZ), 128, 0, stream>>>(QKV, VT, AOT);

  // 7) proj seq linear: OUT2[b][m][c] = sum_n Wp0[m,n]*AO[b,n,c] + bp0[m]
  gemm_bf16<<<dim3(SEQ / 128, CH / 64, BSZ), 256, 0, stream>>>(
      Wp0b, SEQ, 0, AOT, SEQ, (long long)CH * SEQ,
      OUT2, CH, (long long)SEQ * CH, 0, bp0, nullptr, SEQ);

  // 8) proj channel linear (f32 out): out[b][n][d] = sum_c OUT2[b,n,c]*Wp1[d,c] + bp1[d]
  gemm_bf16<<<dim3(SEQ / 128, CH / 64, BSZ), 256, 0, stream>>>(
      OUT2, CH, (long long)SEQ * CH, Wp1b, CH, 0,
      out, CH, (long long)SEQ * CH, 1, nullptr, bp1, CH);
}
